// MultiViewLoss_86887188398391
// MI455X (gfx1250) — matrix operand order repaired and checked by reading, not compiled, not run on hardware
//
#include <hip/hip_runtime.h>

// ---------------------------------------------------------------------------
// MI455X (gfx1250) nearest-neighbor margin loss.
// Strategy: fused GEMM+row-min using v_wmma_f32_16x16x32_f16 with split-f16
// (hi+lo) operands for near-fp32 accuracy at f16 matrix throughput.
// ---------------------------------------------------------------------------

typedef __attribute__((ext_vector_type(16))) _Float16 v16h;
typedef __attribute__((ext_vector_type(8)))  _Float16 h8;
typedef __attribute__((ext_vector_type(8)))  float    v8f;

constexpr int kN1 = 16384;            // rows of view1
constexpr int kN2 = 16384;            // rows of view2
constexpr int kC  = 512;              // feature dim (K)
constexpr int BM  = 128;              // A rows per workgroup
constexpr int BN  = 128;              // B rows (output cols) per inner tile
constexpr int BK  = 32;               // K per WMMA
constexpr int KT  = kC / BK;          // 16 k-steps
constexpr int NSPLIT = 8;             // view2 chunks (grid.y)
constexpr int CHUNK  = kN2 / NSPLIT;  // 2048
constexpr int NTILES = CHUNK / BN;    // 16
constexpr int LDA = kC + 8;           // padded LDS row stride (A), bank-safe
constexpr int LDB = BK + 8;           // padded LDS row stride (B slice)
constexpr int A_ELEMS = BM * LDA;     // 66560 f16
constexpr int B_ELEMS = BM * LDB;     // 5120 f16 (one buffer, one of hi/lo)
constexpr int SMEM_ELEMS = 2 * A_ELEMS + 4 * B_ELEMS;   // 153600 f16
constexpr size_t SMEM_BYTES = (size_t)SMEM_ELEMS * 2;   // 307200 B <= 320KB

struct F16x4 { float4 x0, x1, x2, x3; };

__device__ __forceinline__ F16x4 load16(const float* __restrict__ p) {
  F16x4 r;
  r.x0 = *reinterpret_cast<const float4*>(p + 0);
  r.x1 = *reinterpret_cast<const float4*>(p + 4);
  r.x2 = *reinterpret_cast<const float4*>(p + 8);
  r.x3 = *reinterpret_cast<const float4*>(p + 12);
  return r;
}

// Split 16 fp32 values into f16 hi + f16 lo residual and store to LDS.
__device__ __forceinline__ void split_store16(const F16x4& rg,
                                              _Float16* hi, _Float16* lo) {
  float f[16] = { rg.x0.x, rg.x0.y, rg.x0.z, rg.x0.w,
                  rg.x1.x, rg.x1.y, rg.x1.z, rg.x1.w,
                  rg.x2.x, rg.x2.y, rg.x2.z, rg.x2.w,
                  rg.x3.x, rg.x3.y, rg.x3.z, rg.x3.w };
  h8 vh0, vh1, vl0, vl1;
#pragma unroll
  for (int i = 0; i < 8; ++i) {
    float x = f[i];
    _Float16 hh = (_Float16)x;
    vh0[i] = hh;
    vl0[i] = (_Float16)(x - (float)hh);
  }
#pragma unroll
  for (int i = 0; i < 8; ++i) {
    float x = f[8 + i];
    _Float16 hh = (_Float16)x;
    vh1[i] = hh;
    vl1[i] = (_Float16)(x - (float)hh);
  }
  *reinterpret_cast<h8*>(hi)     = vh0;
  *reinterpret_cast<h8*>(hi + 8) = vh1;
  *reinterpret_cast<h8*>(lo)     = vl0;
  *reinterpret_cast<h8*>(lo + 8) = vl1;
}

// A fragment (16x32 f16, ISA 7.12.2): lane holds K = lh*8..+7 and 16+lh*8..+7.
__device__ __forceinline__ v16h frag_a(const _Float16* p) {
  union { v16h v; h8 h[2]; } u;
  u.h[0] = *reinterpret_cast<const h8*>(p);
  u.h[1] = *reinterpret_cast<const h8*>(p + 16);
  return u.v;
}

// B fragment (32x16 f16): lane n holds K = lh*8..+7 and 16+lh*8..+7.
__device__ __forceinline__ v16h frag_b(const _Float16* p) {
  union { v16h v; h8 h[2]; } u;
  u.h[0] = *reinterpret_cast<const h8*>(p);
  u.h[1] = *reinterpret_cast<const h8*>(p + 16);
  return u.v;
}

__device__ __forceinline__ v8f wmma_f16(v16h a, v16h b, v8f c) {
  return __builtin_amdgcn_wmma_f32_16x16x32_f16(
      /*neg_a=*/false, a, /*neg_b=*/false, b,
      /*c_mod=*/(short)0, c, /*reuse_a=*/false, /*reuse_b=*/false);
}

// ---------------------------------------------------------------------------
// Kernel 1: squared row norms of both matrices. One wave32 per row.
// ---------------------------------------------------------------------------
__global__ __launch_bounds__(256) void sqnorm_kernel(
    const float* __restrict__ A, const float* __restrict__ B,
    float* __restrict__ sq) {
  const int lane = threadIdx.x & 31;
  const int wave = threadIdx.x >> 5;
  const int row  = blockIdx.x * 8 + wave;          // 0 .. kN1+kN2-1
  const float* src = (row < kN1) ? (A + (size_t)row * kC)
                                 : (B + (size_t)(row - kN1) * kC);
  float s = 0.0f;
#pragma unroll 4
  for (int c = lane; c < kC; c += 32) {
    float x = src[c];
    s = __builtin_fmaf(x, x, s);
  }
#pragma unroll
  for (int off = 16; off > 0; off >>= 1) s += __shfl_xor(s, off, 32);
  if (lane == 0) sq[row] = s;
}

// ---------------------------------------------------------------------------
// Kernel 2: fused split-f16 WMMA GEMM + running min of (||b||^2 - 2 a.b).
// grid = (kN1/BM row tiles, NSPLIT view2 chunks), 256 threads = 8 waves.
// ---------------------------------------------------------------------------
__global__ __launch_bounds__(256) void nnmin_gemm_kernel(
    const float* __restrict__ A, const float* __restrict__ B,
    const float* __restrict__ sq2, float* __restrict__ partial) {
  extern __shared__ char smem_raw[];
  _Float16* sA_hi = reinterpret_cast<_Float16*>(smem_raw);
  _Float16* sA_lo = sA_hi + A_ELEMS;
  _Float16* sB    = sA_lo + A_ELEMS;     // layout: [buf0 hi][buf0 lo][buf1 hi][buf1 lo]

  const int t         = threadIdx.x;
  const int rowBase   = blockIdx.x * BM;
  const int chunkBase = blockIdx.y * CHUNK;

  // ---- Stage the full-K A tile (BM x 512) as split hi/lo f16, once ----
  {
    const int r  = t >> 1;
    const int cg = (t & 1) << 4;
    const float* gp = A + (size_t)(rowBase + r) * kC + cg;
    _Float16* ph = sA_hi + r * LDA + cg;
    _Float16* pl = sA_lo + r * LDA + cg;
#pragma unroll 4
    for (int kb = 0; kb < KT; ++kb) {
      F16x4 rg = load16(gp + kb * BK);
      split_store16(rg, ph + kb * BK, pl + kb * BK);
    }
  }
  __syncthreads();

  const int lane = t & 31;
  const int wave = t >> 5;
  const int mw = wave >> 1;        // 0..3 : which 32-row M strip
  const int nw = wave & 1;         // 0..1 : which 64-col N strip
  const int ln = lane & 15;
  const int lh = lane >> 4;

  const int aoff0 = (mw * 32 + ln) * LDA + lh * 8;   // A frag base (Mtile 0)
  const int aoff1 = aoff0 + 16 * LDA;                // Mtile 1
  const int boff  = (nw * 64 + ln) * LDB + lh * 8;   // B frag base (Ntile 0)

  const int br  = t >> 1;          // B staging: 2 threads per slice row
  const int bcg = (t & 1) << 4;

  float vmin[2][8];
#pragma unroll
  for (int i = 0; i < 2; ++i)
#pragma unroll
    for (int r = 0; r < 8; ++r) vmin[i][r] = 3.0e38f;

  for (int nt = 0; nt < NTILES; ++nt) {
    const int colBase = chunkBase + nt * BN;

    v8f acc[2][4];
#pragma unroll
    for (int i = 0; i < 2; ++i)
#pragma unroll
      for (int j = 0; j < 4; ++j)
#pragma unroll
        for (int r = 0; r < 8; ++r) acc[i][j][r] = 0.0f;

    const float* bsrc = B + (size_t)(colBase + br) * kC + bcg;

    // prologue: stage k-slice 0 into buffer 0
    F16x4 rg = load16(bsrc);
    split_store16(rg, sB + br * LDB + bcg, sB + B_ELEMS + br * LDB + bcg);
    __syncthreads();

    for (int kt = 0; kt < KT; ++kt) {
      if (kt + 1 < KT) rg = load16(bsrc + (kt + 1) * BK);   // prefetch next slice

      const _Float16* bh = sB + (kt & 1) * (2 * B_ELEMS);
      const _Float16* bl = bh + B_ELEMS;
      const int ak = kt * BK;

      v16h a0h = frag_a(sA_hi + aoff0 + ak);
      v16h a0l = frag_a(sA_lo + aoff0 + ak);
      v16h a1h = frag_a(sA_hi + aoff1 + ak);
      v16h a1l = frag_a(sA_lo + aoff1 + ak);

#pragma unroll
      for (int j = 0; j < 4; ++j) {
        v16h bhj = frag_b(bh + boff + j * 16 * LDB);
        v16h blj = frag_b(bl + boff + j * 16 * LDB);
        // split product: hi*hi + hi*lo + lo*hi (lo*lo negligible)
        acc[0][j] = wmma_f16(a0h, bhj, acc[0][j]);
        acc[1][j] = wmma_f16(a1h, bhj, acc[1][j]);
        acc[0][j] = wmma_f16(a0h, blj, acc[0][j]);
        acc[1][j] = wmma_f16(a1h, blj, acc[1][j]);
        acc[0][j] = wmma_f16(a0l, bhj, acc[0][j]);
        acc[1][j] = wmma_f16(a1l, bhj, acc[1][j]);
      }

      if (kt + 1 < KT) {           // store prefetched slice into other buffer
        _Float16* dh = sB + ((kt + 1) & 1) * (2 * B_ELEMS) + br * LDB + bcg;
        split_store16(rg, dh, dh + B_ELEMS);
      }
      __syncthreads();
    }

    // epilogue: fold (||b_m||^2 - 2 * inner) into per-row running min.
    // C/D layout: element r of lane L -> row (r + 8*lh), col ln.
#pragma unroll
    for (int j = 0; j < 4; ++j) {
      const float s = sq2[colBase + nw * 64 + j * 16 + ln];
#pragma unroll
      for (int i = 0; i < 2; ++i)
#pragma unroll
        for (int r = 0; r < 8; ++r)
          vmin[i][r] = fminf(vmin[i][r], __builtin_fmaf(-2.0f, acc[i][j][r], s));
    }
  }

  // min across the 16 lanes sharing a row (wave32: xor 1,2,4,8 stays in group)
  float* ldsmin = reinterpret_cast<float*>(sB);   // reuse B buffers (done with them)
#pragma unroll
  for (int i = 0; i < 2; ++i)
#pragma unroll
    for (int r = 0; r < 8; ++r) {
      float v = vmin[i][r];
#pragma unroll
      for (int off = 1; off < 16; off <<= 1) v = fminf(v, __shfl_xor(v, off, 32));
      if (ln == 0) ldsmin[nw * BM + mw * 32 + i * 16 + lh * 8 + r] = v;
    }
  __syncthreads();
  if (t < BM) {
    float v = fminf(ldsmin[t], ldsmin[BM + t]);   // combine the two N-waves
    partial[(size_t)(rowBase + t) * NSPLIT + blockIdx.y] = v;
  }
}

// ---------------------------------------------------------------------------
// Kernel 3: combine chunk mins, sqrt/relu/mean. Single block => deterministic.
// ---------------------------------------------------------------------------
__global__ __launch_bounds__(256) void finalize_kernel(
    const float* __restrict__ sq1, const float* __restrict__ partial,
    float* __restrict__ out) {
  __shared__ float red[256];
  float s = 0.0f;
  for (int r = threadIdx.x; r < kN1; r += 256) {
    const float* p = partial + (size_t)r * NSPLIT;
    float mn = p[0];
#pragma unroll
    for (int k = 1; k < NSPLIT; ++k) mn = fminf(mn, p[k]);
    float d2 = fmaxf(sq1[r] + mn, 0.0f);      // clamp commutes with min
    float d  = sqrtf(d2);
    s += fmaxf(d - 0.1f, 0.0f);
  }
  red[threadIdx.x] = s;
  __syncthreads();
  for (int st = 128; st > 0; st >>= 1) {
    if ((int)threadIdx.x < st) red[threadIdx.x] += red[threadIdx.x + st];
    __syncthreads();
  }
  if (threadIdx.x == 0) out[0] = red[0] * (1.0f / (float)kN1);
}

// ---------------------------------------------------------------------------
extern "C" void kernel_launch(void* const* d_in, const int* in_sizes, int n_in,
                              void* d_out, int out_size, void* d_ws, size_t ws_size,
                              hipStream_t stream) {
  (void)in_sizes; (void)n_in; (void)out_size; (void)ws_size;
  const float* A = (const float*)d_in[0];   // view1 [16384][512] f32
  const float* B = (const float*)d_in[1];   // view2 [16384][512] f32
  float* ws      = (float*)d_ws;
  float* sq      = ws;                      // [kN1 + kN2] row norms (sq1 then sq2)
  float* partial = ws + (kN1 + kN2);        // [kN1][NSPLIT] chunk mins
  float* out     = (float*)d_out;

  sqnorm_kernel<<<(kN1 + kN2) / 8, 256, 0, stream>>>(A, B, sq);

  dim3 grid(kN1 / BM, NSPLIT);
  nnmin_gemm_kernel<<<grid, 256, SMEM_BYTES, stream>>>(A, B, sq + kN1, partial);

  finalize_kernel<<<1, 256, 0, stream>>>(sq, partial, out);
}
